// ACIE_Core_27384711479982
// MI455X (gfx1250) — compile-verified, matrix operand order checked by reading, not run on hardware
//
#include <hip/hip_runtime.h>
#include <hip/hip_bf16.h>
#include <stddef.h>

typedef __bf16 bf16;
typedef __attribute__((ext_vector_type(16))) __bf16 v16bf;
typedef __attribute__((ext_vector_type(8)))  __bf16 v8bf;
typedef __attribute__((ext_vector_type(8)))  float  v8f;
typedef __attribute__((ext_vector_type(2)))  float  v2f;

#define NBINS 100

// ---------------------------------------------------------------------------
// Kernel 1: per-row histogram + entropy gate. One 256-thread block per row.
// float4 streaming loads; per-wave private histograms to reduce LDS atomic
// serialization; two tree reductions for total count and entropy.
// ---------------------------------------------------------------------------
__global__ __launch_bounds__(256) void hist_gate_kernel(
    const float* __restrict__ ev, float* __restrict__ gate, int D)
{
  __shared__ unsigned bins[8][NBINS];
  __shared__ float red[256];
  const int b = blockIdx.x;
  const int tid = threadIdx.x;
  const int wave = tid >> 5;
  const float4* row = (const float4*)(ev + (size_t)b * D);

  for (int i = tid; i < 8 * NBINS; i += 256) (&bins[0][0])[i] = 0u;
  __syncthreads();

  const int D4 = D >> 2;
  for (int i = tid; i < D4; i += 256) {
    float4 v = row[i];
    float xs[4] = {v.x, v.y, v.z, v.w};
    #pragma unroll
    for (int j = 0; j < 4; ++j) {
      float x = xs[j];
      if (x >= -5.0f && x <= 5.0f) {
        int idx = (int)floorf((x + 5.0f) * 10.0f);
        idx = idx < 0 ? 0 : (idx > NBINS - 1 ? NBINS - 1 : idx);
        atomicAdd(&bins[wave][idx], 1u);
      }
    }
  }
  __syncthreads();

  float c = 0.0f;
  if (tid < NBINS) {
    unsigned s = 0;
    #pragma unroll
    for (int w = 0; w < 8; ++w) s += bins[w][tid];
    c = (float)s;
  }
  red[tid] = c;
  __syncthreads();
  for (int s = 128; s > 0; s >>= 1) {
    if (tid < s) red[tid] += red[tid + s];
    __syncthreads();
  }
  const float total = red[0];
  __syncthreads();

  float ent = 0.0f;
  if (tid < NBINS) {
    float p = c / (total + 1e-9f);
    ent = -p * logf(p + 1e-9f);
  }
  red[tid] = ent;
  __syncthreads();
  for (int s = 128; s > 0; s >>= 1) {
    if (tid < s) red[tid] += red[tid + s];
    __syncthreads();
  }
  if (tid == 0) gate[b] = (red[0] >= 2.5f) ? 1.0f : 0.0f;
}

// ---------------------------------------------------------------------------
// Generic NT bf16 WMMA GEMM: C[M,N] = A[M,K] x B[N,K]^T, K-contig operands.
// BIG config: 128x128 block, 8 waves as 4x2, wave tile 32x64 (8 accums).
// SMALL config: 128x64 block, 8 waves x (16x64 strip).
// Double-buffered LDS. Pure-bf16 operand modes use CDNA5 async-to-LDS copies
// (ASYNCcnt); f32 operands stage through registers with on-the-fly bf16 cvt.
// MODE epilogues:
//  0: filtered = gate[row] * acc             -> bf16
//  1: state    = acc                         -> bf16
//  2: t        = (acc+b1[c] > 0) ? vsum[c]:0 -> bf16
//  3: s_adv    = state_in - 0.1*sign(acc)    -> bf16
//  4: h2       = relu(acc + b1[c])           -> bf16
//  5: logits   = acc + b2[c]                 -> f32
// ---------------------------------------------------------------------------
union FragBF { v16bf v; v8bf h[2]; };

__device__ __forceinline__ void async_copy_b128(const void* gptr, void* lptr) {
  unsigned lds = (unsigned)(unsigned long long)
      (__attribute__((address_space(3))) void*)lptr;
  unsigned long long gaddr = (unsigned long long)(size_t)gptr;
  asm volatile("global_load_async_to_lds_b128 %0, %1, off"
               :: "v"(lds), "v"(gaddr) : "memory");
}
__device__ __forceinline__ void wait_async0() {
  asm volatile("s_wait_asynccnt 0x0" ::: "memory");
}

template<int MODE, bool AF32, bool BF32, bool BIG>
__global__ __launch_bounds__(256) void gemm_nt(
    const void* __restrict__ Aptr, const void* __restrict__ Bptr,
    int M, int N, int K,
    const float* __restrict__ gate,
    const bf16*  __restrict__ state_in,
    const float* __restrict__ bias,
    const float* __restrict__ vsum,
    void* __restrict__ Cptr)
{
  constexpr int  BN    = BIG ? 128 : 64;
  constexpr int  MT    = BIG ? 2 : 1;                  // 16-row tiles per wave
  constexpr bool ASYNC = (!AF32) && (!BF32);
  constexpr int  NCA   = AF32 ? 4 : 2;                 // A chunks / thread
  constexpr int  NCB   = (BF32 ? (BN * 8) : (BN * 4)) / 256;

  __shared__ alignas(16) bf16 As[2][128 * 40];  // row stride 40 bf16 = 80B
  __shared__ alignas(16) bf16 Bs[2][BN * 40];

  const int tid  = threadIdx.x;
  const int wave = tid >> 5;
  const int lane = tid & 31;
  const int bm   = blockIdx.y * 128;     // y = M so N-blocks sharing A are adjacent
  const int bn   = blockIdx.x * BN;
  const int nlane = lane & 15;
  const int kpart = (lane >> 4) * 8;
  const int wr = BIG ? (wave & 3) : wave;        // wave row index
  const int wc = BIG ? (wave >> 2) : 0;          // wave col index

  v8f acc[MT][4];
  #pragma unroll
  for (int m = 0; m < MT; ++m)
    #pragma unroll
    for (int t = 0; t < 4; ++t)
      #pragma unroll
      for (int e = 0; e < 8; ++e) acc[m][t][e] = 0.0f;

  auto computeTile = [&](int p) {
    FragBF af[MT];
    #pragma unroll
    for (int m = 0; m < MT; ++m) {
      const int ar = wr * (16 * MT) + m * 16 + nlane;
      af[m].h[0] = *(const v8bf*)&As[p][ar * 40 + kpart];
      af[m].h[1] = *(const v8bf*)&As[p][ar * 40 + kpart + 16];
    }
    #pragma unroll
    for (int t = 0; t < 4; ++t) {
      FragBF bfr;
      const int br = wc * 64 + t * 16 + nlane;
      bfr.h[0] = *(const v8bf*)&Bs[p][br * 40 + kpart];
      bfr.h[1] = *(const v8bf*)&Bs[p][br * 40 + kpart + 16];
      #pragma unroll
      for (int m = 0; m < MT; ++m)
        acc[m][t] = __builtin_amdgcn_wmma_f32_16x16x32_bf16(
            false, af[m].v, false, bfr.v, (short)0, acc[m][t], false, false);
    }
  };

  if constexpr (ASYNC) {
    const bf16* A = (const bf16*)Aptr;
    const bf16* Bg = (const bf16*)Bptr;
    auto issue = [&](int kb, int p) {
      #pragma unroll
      for (int i = 0; i < NCA; ++i) {
        int c = tid + 256 * i, r = c >> 2, kc = c & 3;
        async_copy_b128(A + (size_t)(bm + r) * K + kb + kc * 8,
                        &As[p][r * 40 + kc * 8]);
      }
      #pragma unroll
      for (int i = 0; i < NCB; ++i) {
        int c = tid + 256 * i, r = c >> 2, kc = c & 3;
        async_copy_b128(Bg + (size_t)(bn + r) * K + kb + kc * 8,
                        &Bs[p][r * 40 + kc * 8]);
      }
    };
    issue(0, 0);
    wait_async0();
    __syncthreads();
    int p = 0;
    for (int kb = 0; kb < K; kb += 32) {
      const bool more = (kb + 32) < K;
      if (more) issue(kb + 32, 1 - p);
      computeTile(p);
      if (more) wait_async0();
      __syncthreads();
      if (more) p ^= 1;
    }
  } else {
    float4 raf[4]; v8bf rah[2];
    float4 rbf[4]; v8bf rbh[2];
    auto loadA = [&](int kb) {
      if constexpr (AF32) {
        const float* A = (const float*)Aptr;
        #pragma unroll
        for (int i = 0; i < 4; ++i) {
          int c = tid + 256 * i, r = c >> 3, kc = c & 7;
          raf[i] = *(const float4*)(A + (size_t)(bm + r) * K + kb + kc * 4);
        }
      } else {
        const bf16* A = (const bf16*)Aptr;
        #pragma unroll
        for (int i = 0; i < 2; ++i) {
          int c = tid + 256 * i, r = c >> 2, kc = c & 3;
          rah[i] = *(const v8bf*)(A + (size_t)(bm + r) * K + kb + kc * 8);
        }
      }
    };
    auto loadB = [&](int kb) {
      if constexpr (BF32) {
        const float* Bg = (const float*)Bptr;
        #pragma unroll
        for (int i = 0; i < NCB; ++i) {
          int c = tid + 256 * i, r = c >> 3, kc = c & 7;
          rbf[i] = *(const float4*)(Bg + (size_t)(bn + r) * K + kb + kc * 4);
        }
      } else {
        const bf16* Bg = (const bf16*)Bptr;
        #pragma unroll
        for (int i = 0; i < NCB; ++i) {
          int c = tid + 256 * i, r = c >> 2, kc = c & 3;
          rbh[i] = *(const v8bf*)(Bg + (size_t)(bn + r) * K + kb + kc * 8);
        }
      }
    };
    auto storeA = [&](int p) {
      if constexpr (AF32) {
        #pragma unroll
        for (int i = 0; i < 4; ++i) {
          int c = tid + 256 * i, r = c >> 3, kc = c & 7;
          bf16* d = &As[p][r * 40 + kc * 4];
          d[0] = (bf16)raf[i].x; d[1] = (bf16)raf[i].y;
          d[2] = (bf16)raf[i].z; d[3] = (bf16)raf[i].w;
        }
      } else {
        #pragma unroll
        for (int i = 0; i < 2; ++i) {
          int c = tid + 256 * i, r = c >> 2, kc = c & 3;
          *(v8bf*)&As[p][r * 40 + kc * 8] = rah[i];
        }
      }
    };
    auto storeB = [&](int p) {
      if constexpr (BF32) {
        #pragma unroll
        for (int i = 0; i < NCB; ++i) {
          int c = tid + 256 * i, r = c >> 3, kc = c & 7;
          bf16* d = &Bs[p][r * 40 + kc * 4];
          d[0] = (bf16)rbf[i].x; d[1] = (bf16)rbf[i].y;
          d[2] = (bf16)rbf[i].z; d[3] = (bf16)rbf[i].w;
        }
      } else {
        #pragma unroll
        for (int i = 0; i < NCB; ++i) {
          int c = tid + 256 * i, r = c >> 2, kc = c & 3;
          *(v8bf*)&Bs[p][r * 40 + kc * 8] = rbh[i];
        }
      }
    };

    loadA(0); loadB(0);
    storeA(0); storeB(0);
    int p = 0;
    for (int kb = 0; kb < K; kb += 32) {
      const bool more = (kb + 32) < K;
      if (more) { loadA(kb + 32); loadB(kb + 32); }  // overlap with compute
      __syncthreads();
      computeTile(p);
      if (more) { storeA(1 - p); storeB(1 - p); p ^= 1; }
    }
  }

  // epilogue: VGPR v, lanes<16 -> M=v, lanes>=16 -> M=v+8
  const int hrow = (lane >> 4) * 8;
  #pragma unroll
  for (int m = 0; m < MT; ++m) {
    #pragma unroll
    for (int t = 0; t < 4; ++t) {
      #pragma unroll
      for (int v = 0; v < 8; ++v) {
        const int row = bm + wr * (16 * MT) + m * 16 + v + hrow;
        const int col = bn + wc * 64 + t * 16 + nlane;
        const float val = acc[m][t][v];
        const size_t off = (size_t)row * N + col;
        if (MODE == 0) {
          ((bf16*)Cptr)[off] = (bf16)(val * gate[row]);
        } else if (MODE == 1) {
          ((bf16*)Cptr)[off] = (bf16)val;
        } else if (MODE == 2) {
          ((bf16*)Cptr)[off] = (bf16)(((val + bias[col]) > 0.0f) ? vsum[col] : 0.0f);
        } else if (MODE == 3) {
          float s = (val > 0.0f) ? 0.1f : ((val < 0.0f) ? -0.1f : 0.0f);
          ((bf16*)Cptr)[off] = (bf16)((float)state_in[off] - s);
        } else if (MODE == 4) {
          float z = val + bias[col];
          ((bf16*)Cptr)[off] = (bf16)(z > 0.0f ? z : 0.0f);
        } else {
          ((float*)Cptr)[off] = val + bias[col];
        }
      }
    }
  }
}

// ---------------------------------------------------------------------------
// f32 WMMA GEMM for expm Taylor terms: C = P x M (B operand = M^T rows, NT).
// 512x512x512; accumulates coef * trace(C) via diagonal atomics.
// ---------------------------------------------------------------------------
__global__ __launch_bounds__(256) void gemm_expm(
    const float* __restrict__ P, const float* __restrict__ MT,
    float* __restrict__ C, float* __restrict__ traceAcc, float coef)
{
  __shared__ float As[128 * 36];  // row stride 36 f32 = 144B
  __shared__ float Bs[64 * 36];
  const int tid  = threadIdx.x;
  const int wave = tid >> 5;
  const int lane = tid & 31;
  const int bm   = blockIdx.x * 128;
  const int bn   = blockIdx.y * 64;
  const int nlane = lane & 15;
  const int K = 512, N = 512;

  v8f acc[4];
  #pragma unroll
  for (int t = 0; t < 4; ++t)
    #pragma unroll
    for (int e = 0; e < 8; ++e) acc[t][e] = 0.0f;

  for (int kb = 0; kb < K; kb += 32) {
    for (int c = tid; c < 128 * 8; c += 256) {
      int r = c >> 3, kc = c & 7;
      *(float4*)&As[r * 36 + kc * 4] =
          *(const float4*)(P + (size_t)(bm + r) * K + kb + kc * 4);
    }
    for (int c = tid; c < 64 * 8; c += 256) {
      int r = c >> 3, kc = c & 7;
      *(float4*)&Bs[r * 36 + kc * 4] =
          *(const float4*)(MT + (size_t)(bn + r) * K + kb + kc * 4);
    }
    __syncthreads();

    const int arow = wave * 16 + nlane;
    #pragma unroll
    for (int ko = 0; ko < 8; ++ko) {
      const int ks = ko * 4 + (lane >> 4) * 2;
      v2f a = *(const v2f*)&As[arow * 36 + ks];
      #pragma unroll
      for (int t = 0; t < 4; ++t) {
        v2f b = *(const v2f*)&Bs[(t * 16 + nlane) * 36 + ks];
        acc[t] = __builtin_amdgcn_wmma_f32_16x16x4_f32(
            false, a, false, b, (short)0, acc[t], false, false);
      }
    }
    __syncthreads();
  }

  const int hrow = (lane >> 4) * 8;
  #pragma unroll
  for (int t = 0; t < 4; ++t) {
    #pragma unroll
    for (int v = 0; v < 8; ++v) {
      const int row = bm + wave * 16 + v + hrow;
      const int col = bn + t * 16 + nlane;
      const float val = acc[t][v];
      C[(size_t)row * N + col] = val;
      if (row == col) atomicAdd(traceAcc, val * coef);
    }
  }
}

// ---------------------------------------------------------------------------
// Small prep / finalize kernels
// ---------------------------------------------------------------------------
__global__ void init_trace_kernel(float* trace) {
  if (blockIdx.x == 0 && threadIdx.x == 0) trace[0] = 0.0f;
}

__global__ __launch_bounds__(256) void prep_adj_kernel(
    const float* __restrict__ adj, float* __restrict__ Mf, float* __restrict__ MT,
    bf16* __restrict__ adjT, float* __restrict__ traceAcc)
{
  const int i = blockIdx.x * 256 + threadIdx.x;     // 512*512
  const int r = i >> 9, c = i & 511;
  const float a = adj[i];
  const float m = a * a;
  Mf[i] = m;
  MT[(size_t)c * 512 + r] = m;
  adjT[(size_t)c * 512 + r] = (bf16)a;
  if (r == c) atomicAdd(traceAcc, m);               // k=1 Taylor term
}

__global__ __launch_bounds__(256) void prep_w_kernel(
    const float* __restrict__ w1, const float* __restrict__ w2,
    bf16* __restrict__ w1T, bf16* __restrict__ w2T, float* __restrict__ vsum)
{
  const int i = blockIdx.x * 256 + threadIdx.x;
  if (i < 512 * 128) { int n = i >> 7, h = i & 127; w1T[(size_t)h * 512 + n] = (bf16)w1[i]; }
  if (i < 128 * 64)  { int h = i >> 6, a = i & 63;  w2T[(size_t)a * 128 + h] = (bf16)w2[i]; }
  if (i < 128) {
    float s = 0.0f;
    for (int a = 0; a < 64; ++a) s += w2[i * 64 + a];
    vsum[i] = s;
  }
}

__global__ __launch_bounds__(256) void softmax_kernel(
    const float* __restrict__ logits, float* __restrict__ out)
{
  const int wave = threadIdx.x >> 5;
  const int lane = threadIdx.x & 31;
  const int row  = blockIdx.x * 8 + wave;
  const float* L = logits + (size_t)row * 64;
  float x0 = L[lane], x1 = L[lane + 32];
  float mx = fmaxf(x0, x1);
  for (int o = 16; o > 0; o >>= 1) mx = fmaxf(mx, __shfl_xor(mx, o, 32));
  float e0 = expf(x0 - mx), e1 = expf(x1 - mx);
  float s = e0 + e1;
  for (int o = 16; o > 0; o >>= 1) s += __shfl_xor(s, o, 32);
  float inv = 1.0f / s;
  float* O = out + (size_t)row * 64;
  O[lane] = e0 * inv;
  O[lane + 32] = e1 * inv;
}

__global__ void write_trace_kernel(const float* __restrict__ trace, float* __restrict__ dst) {
  if (blockIdx.x == 0 && threadIdx.x == 0) dst[0] = trace[0];
}

// ---------------------------------------------------------------------------
// Host launcher
// ---------------------------------------------------------------------------
extern "C" void kernel_launch(void* const* d_in, const int* in_sizes, int n_in,
                              void* d_out, int out_size, void* d_ws, size_t ws_size,
                              hipStream_t stream) {
  const int B = 16384, D = 5120, N = 512, H = 128, A = 64;
  const float* ev   = (const float*)d_in[0];
  const float* sens = (const float*)d_in[1];
  const float* adj  = (const float*)d_in[2];
  const float* w1   = (const float*)d_in[3];
  const float* b1   = (const float*)d_in[4];
  const float* w2   = (const float*)d_in[5];
  const float* b2   = (const float*)d_in[6];
  float* out = (float*)d_out;

  size_t off = 0;
  auto carve = [&](size_t bytes) -> void* {
    off = (off + 255) & ~(size_t)255;
    void* p = (char*)d_ws + off;
    off += bytes;
    return p;
  };
  float* gate     = (float*)carve((size_t)B * 4);
  bf16*  filtered = (bf16*) carve((size_t)B * N * 2);
  bf16*  state    = (bf16*) carve((size_t)B * N * 2);
  bf16*  tbuf     = (bf16*) carve((size_t)B * H * 2);
  bf16*  sadv     = (bf16*) carve((size_t)B * N * 2);
  bf16*  h2       = (bf16*) carve((size_t)B * H * 2);
  float* logits   = (float*)carve((size_t)B * A * 4);
  bf16*  adjT     = (bf16*) carve((size_t)N * N * 2);
  float* Mf       = (float*)carve((size_t)N * N * 4);
  float* Mt       = (float*)carve((size_t)N * N * 4);
  float* Pa       = (float*)carve((size_t)N * N * 4);
  float* Pb       = (float*)carve((size_t)N * N * 4);
  bf16*  w1T      = (bf16*) carve((size_t)H * N * 2);
  bf16*  w2T      = (bf16*) carve((size_t)A * H * 2);
  float* vsum     = (float*)carve((size_t)H * 4);
  float* trace    = (float*)carve(256);

  // prep
  init_trace_kernel<<<1, 1, 0, stream>>>(trace);
  prep_adj_kernel<<<(N * N) / 256, 256, 0, stream>>>(adj, Mf, Mt, adjT, trace);
  prep_w_kernel<<<(N * H) / 256, 256, 0, stream>>>(w1, w2, w1T, w2T, vsum);
  hist_gate_kernel<<<B, 256, 0, stream>>>(ev, gate, D);

  // Layer 1+2: compressed (gated) and causal state. grid = (N/128, M/128)
  gemm_nt<0, true,  true,  true><<<dim3(N / 128, B / 128), 256, 0, stream>>>(
      ev, sens, B, N, D, gate, nullptr, nullptr, nullptr, filtered);
  gemm_nt<1, false, false, true><<<dim3(N / 128, B / 128), 256, 0, stream>>>(
      filtered, adjT, B, N, N, nullptr, nullptr, nullptr, nullptr, state);

  // Layer 3: analytic FGSM grad sign + robust policy
  gemm_nt<2, false, false, true><<<dim3(H / 128, B / 128), 256, 0, stream>>>(
      state, w1T, B, H, N, nullptr, nullptr, b1, vsum, tbuf);
  gemm_nt<3, false, true,  true><<<dim3(N / 128, B / 128), 256, 0, stream>>>(
      tbuf, w1, B, N, H, nullptr, state, nullptr, nullptr, sadv);
  gemm_nt<4, false, false, true><<<dim3(H / 128, B / 128), 256, 0, stream>>>(
      sadv, w1T, B, H, N, nullptr, nullptr, b1, nullptr, h2);
  gemm_nt<5, false, false, false><<<dim3(A / 64, B / 128), 256, 0, stream>>>(
      h2, w2T, B, A, H, nullptr, nullptr, b2, nullptr, logits);

  // NOTEARS: trace(expm(A*A)) - n  ==  sum_{k>=1} tr(M^k)/k!
  double fact = 1.0;
  for (int k = 2; k <= 24; ++k) {
    fact *= (double)k;
    const float coef = (float)(1.0 / fact);
    const float* prev = (k == 2) ? Mf : ((k % 2 == 0) ? Pb : Pa);
    float* next = (k % 2 == 0) ? Pa : Pb;
    gemm_expm<<<dim3(N / 128, N / 64), 256, 0, stream>>>(prev, Mt, next, trace, coef);
  }

  // outputs
  softmax_kernel<<<B / 8, 256, 0, stream>>>(logits, out);
  write_trace_kernel<<<1, 1, 0, stream>>>(trace, out + (size_t)B * A);
}
